// EagleAttention_33990371181009
// MI455X (gfx1250) — matrix operand order repaired and checked by reading, not compiled, not run on hardware
//
#include <hip/hip_runtime.h>

// ---------- types ----------
typedef __attribute__((ext_vector_type(16))) __bf16 v16bf;
typedef __attribute__((ext_vector_type(8)))  float  v8f;
typedef __attribute__((ext_vector_type(4)))  unsigned int v4u;

// exact parameter type expected by __builtin_amdgcn_global_load_async_to_lds_b128
typedef int v4i __attribute__((vector_size(16)));
typedef __attribute__((address_space(1))) v4i g_v4i;   // global
typedef __attribute__((address_space(3))) v4i l_v4i;   // LDS

union BF16Frag { v16bf v; v4u q[2]; };

__device__ __forceinline__ unsigned short f2bf(float f) {
  unsigned int u = __float_as_uint(f);
  u += 0x7fffu + ((u >> 16) & 1u);          // round-to-nearest-even
  return (unsigned short)(u >> 16);
}

#define HAS_ASYNC_LDS __has_builtin(__builtin_amdgcn_global_load_async_to_lds_b128)

// ---------- problem constants ----------
constexpr int Hc   = 2304;
constexpr int NHc  = 8;
constexpr int NKVc = 4;
constexpr int Dc   = 256;
constexpr int Bc   = 2;
constexpr int Sc   = 2048;
constexpr int ROWS = Bc * Sc;       // 4096
constexpr int KIN  = 2 * Hc;        // 4608
constexpr int NQ   = NHc  * Dc;     // 2048
constexpr int NKVN = NKVc * Dc;     // 1024

// ---------- elementwise convert f32 -> bf16 ----------
__global__ void conv_bf16_kernel(const float* __restrict__ in,
                                 unsigned short* __restrict__ out, int n) {
  int i = blockIdx.x * blockDim.x + threadIdx.x;
  if (i < n) out[i] = f2bf(in[i]);
}

// ---------- transpose+convert: W[K,N] f32 -> Wt[N,K] bf16 ----------
__global__ void tconv_bf16_kernel(const float* __restrict__ in,
                                  unsigned short* __restrict__ out,
                                  int K, int N) {
  int i = blockIdx.x * blockDim.x + threadIdx.x;
  if (i >= K * N) return;
  int k = i / N, n = i % N;
  out[(size_t)n * K + k] = f2bf(in[i]);
}

// ---------- bf16 GEMM: C[M,N] = A[M,K] * Bt[N,K]^T, f32 accumulate ----------
// 256 threads = 8 waves; wave w -> rows m0 = blockIdx.x*256 + w*32 (32 rows),
// cols n0 = blockIdx.y*64. 8 WMMA per K-step of 32; B-fragments reused by 2 A-fragments.
__global__ __launch_bounds__(256) void gemm_bf16_nt(
    const unsigned short* __restrict__ A,
    const unsigned short* __restrict__ Bt,
    float* __restrict__ C, int M, int N, int K) {
  const int wave = threadIdx.x >> 5;
  const int lane = threadIdx.x & 31;
  const int m    = lane & 15;
  const int half = lane >> 4;
  const int m0 = blockIdx.x * 256 + wave * 32;
  const int n0 = blockIdx.y * 64;

  v8f acc0[4], acc1[4];
#pragma unroll
  for (int t = 0; t < 4; ++t) { acc0[t] = (v8f){}; acc1[t] = (v8f){}; }

  const unsigned short* arow0 = A + (size_t)(m0 + m) * K;
  const unsigned short* arow1 = arow0 + (size_t)16 * K;
  const unsigned short* brow  = Bt + (size_t)(n0 + m) * K;

  for (int k0 = 0; k0 < K; k0 += 32) {
    BF16Frag a0, a1;
    a0.q[0] = *(const v4u*)(arow0 + k0 + half * 8);
    a0.q[1] = *(const v4u*)(arow0 + k0 + 16 + half * 8);
    a1.q[0] = *(const v4u*)(arow1 + k0 + half * 8);
    a1.q[1] = *(const v4u*)(arow1 + k0 + 16 + half * 8);
    const unsigned short* bp = brow + k0 + half * 8;
    BF16Frag b[4];
#pragma unroll
    for (int t = 0; t < 4; ++t) {
      b[t].q[0] = *(const v4u*)(bp + (size_t)(t * 16) * K);
      b[t].q[1] = *(const v4u*)(bp + (size_t)(t * 16) * K + 16);
    }
#pragma unroll
    for (int t = 0; t < 4; ++t) {
      acc0[t] = __builtin_amdgcn_wmma_f32_16x16x32_bf16(false, a0.v, false, b[t].v, (short)0, acc0[t], false, false);
      acc1[t] = __builtin_amdgcn_wmma_f32_16x16x32_bf16(false, a1.v, false, b[t].v, (short)0, acc1[t], false, false);
    }
  }
#pragma unroll
  for (int r = 0; r < 8; ++r) {
    int row = m0 + r + 8 * half;
    float* crow = C + (size_t)row * N + n0 + m;
    crow[0]  = acc0[0][r];
    crow[16] = acc0[1][r];
    crow[32] = acc0[2][r];
    crow[48] = acc0[3][r];
    float* crow1 = crow + (size_t)16 * N;
    crow1[0]  = acc1[0][r];
    crow1[16] = acc1[1][r];
    crow1[32] = acc1[2][r];
    crow1[48] = acc1[3][r];
  }
}

// ---------- fused per-head RMSNorm + RoPE, f32 -> bf16 head-major ----------
// grid (ROWS, nheads), block 256 (= D)
__global__ __launch_bounds__(256) void rms_rope_kernel(
    const float* __restrict__ X,        // [ROWS, nheads*D]
    const float* __restrict__ w,        // [D]
    const int*   __restrict__ pos,      // [ROWS]
    unsigned short* __restrict__ out,   // [B, nheads, S, D] bf16
    int nheads) {
  __shared__ float red[256];
  __shared__ float sh[256];
  const int row = blockIdx.x;
  const int h   = blockIdx.y;
  const int d   = threadIdx.x;

  float x = X[(size_t)row * (nheads * Dc) + h * Dc + d];
  red[d] = x * x;
  __syncthreads();
  for (int s = 128; s > 0; s >>= 1) {
    if (d < s) red[d] += red[d + s];
    __syncthreads();
  }
  float rr = rsqrtf(red[0] * (1.0f / Dc) + 1e-6f);
  float nv = x * rr * (1.0f + w[d]);
  sh[d] = nv;
  __syncthreads();

  int i = d & 127;
  const float LOG_BASE = 9.210340371976184f;             // ln(10000)
  float inv = __expf(-(float)i * (LOG_BASE / 128.0f));
  float ang = (float)pos[row] * inv;
  float sn, cs;
  __sincosf(ang, &sn, &cs);
  float partner = (d < 128) ? sh[d + 128] : sh[d - 128];
  float y = (d < 128) ? (nv * cs - partner * sn) : (nv * cs + partner * sn);

  int b = row / Sc, sidx = row % Sc;
  out[(((size_t)b * nheads + h) * Sc + sidx) * Dc + d] = f2bf(y);
}

// ---------- V: [ROWS, NKV*D] f32 -> Vt[B, NKV, D, S] bf16 ----------
__global__ void vtrans_kernel(const float* __restrict__ Vf,
                              unsigned short* __restrict__ Vt) {
  int i = blockIdx.x * blockDim.x + threadIdx.x;
  if (i >= ROWS * NKVN) return;
  int col = i % NKVN, row = i / NKVN;
  int h = col / Dc, d = col % Dc;
  int b = row / Sc, s = row % Sc;
  Vt[(((size_t)b * NKVc + h) * Dc + d) * Sc + s] = f2bf(Vf[i]);
}

// ---------- flash attention: one wave per (b, h, 16-row q tile) ----------
__global__ __launch_bounds__(32) void attn_kernel(
    const unsigned short* __restrict__ Qh,  // [B,NH,S,D]
    const unsigned short* __restrict__ Kh,  // [B,NKV,S,D]
    const unsigned short* __restrict__ Vt,  // [B,NKV,D,S]
    unsigned short* __restrict__ AO) {      // [ROWS, NH*D]
  __shared__ __align__(16) float          s_sc[16 * 32];
  __shared__ __align__(16) unsigned short s_p[16 * 32];
  __shared__ float s_m[16], s_l[16], s_f[16];
#if HAS_ASYNC_LDS
  // V-block staging: 16 d-tiles x 32 lanes x 32B = 16KB
  __shared__ __align__(16) unsigned short s_v[16 * 32 * 16];
#endif

  const int lane = threadIdx.x;
  const int m    = lane & 15;
  const int half = lane >> 4;
  const int qt = blockIdx.x, h = blockIdx.y, b = blockIdx.z;
  const int q0 = qt * 16;
  const int kvh = h >> 1;                  // NH/NKV == 2

  const unsigned short* Qb = Qh + ((size_t)(b * NHc  + h)   * Sc + q0) * Dc;
  const unsigned short* Kb = Kh +  (size_t)(b * NKVc + kvh) * Sc * Dc;
  const unsigned short* Vb = Vt +  (size_t)(b * NKVc + kvh) * Dc * Sc;

  // Q A-fragments for the 8 k-chunks of D=256
  BF16Frag qf[8];
  const unsigned short* qrow = Qb + m * Dc;
#pragma unroll
  for (int c = 0; c < 8; ++c) {
    qf[c].q[0] = *(const v4u*)(qrow + c * 32 + half * 16);
    qf[c].q[1] = *(const v4u*)(qrow + c * 32 + half * 16 + 8);
  }

  v8f accO[16];
#pragma unroll
  for (int t = 0; t < 16; ++t) accO[t] = (v8f){};

  if (lane < 16) { s_m[lane] = -1e30f; s_l[lane] = 0.0f; s_f[lane] = 1.0f; }
  asm volatile("s_wait_dscnt 0x0" ::: "memory");

  for (int key0 = 0; key0 < q0 + 16; key0 += 32) {
#if HAS_ASYNC_LDS
    // Fence last iteration's ds reads of s_v, then asynchronously stage this
    // block's V tiles into LDS; overlaps with the QK^T WMMAs and softmax below.
    asm volatile("s_wait_dscnt 0x0" ::: "memory");
#pragma unroll
    for (int t = 0; t < 16; ++t) {
      const unsigned short* vp = Vb + (size_t)(t * 16 + m) * Sc + key0 + half * 16;
      unsigned short* lp = &s_v[(t * 32 + lane) * 16];
      // INST_OFFSET applies to both global and LDS addresses (ISA §10 async),
      // so one address pair stages 32B with two b128 issues (offset 0 / 16).
      __builtin_amdgcn_global_load_async_to_lds_b128(
          (g_v4i*)(unsigned long long)vp,
          (l_v4i*)(unsigned int)(unsigned long long)lp,
          0, 0);
      __builtin_amdgcn_global_load_async_to_lds_b128(
          (g_v4i*)(unsigned long long)vp,
          (l_v4i*)(unsigned int)(unsigned long long)lp,
          16, 0);
    }
#endif
    // --- scores: 16x32 via 2 C tiles, K-dim 256 ---
    v8f s0 = {}, s1 = {};
    const unsigned short* kp0 = Kb + (size_t)(key0 + m) * Dc;
    const unsigned short* kp1 = kp0 + (size_t)16 * Dc;
#pragma unroll
    for (int c = 0; c < 8; ++c) {
      BF16Frag kf0, kf1;
      kf0.q[0] = *(const v4u*)(kp0 + c * 32 + half * 16);
      kf0.q[1] = *(const v4u*)(kp0 + c * 32 + half * 16 + 8);
      kf1.q[0] = *(const v4u*)(kp1 + c * 32 + half * 16);
      kf1.q[1] = *(const v4u*)(kp1 + c * 32 + half * 16 + 8);
      s0 = __builtin_amdgcn_wmma_f32_16x16x32_bf16(false, qf[c].v, false, kf0.v, (short)0, s0, false, false);
      s1 = __builtin_amdgcn_wmma_f32_16x16x32_bf16(false, qf[c].v, false, kf1.v, (short)0, s1, false, false);
    }
    // scale + causal mask, stage to LDS (C layout: row r+8*half, col m)
#pragma unroll
    for (int r = 0; r < 8; ++r) {
      int row = q0 + r + 8 * half;
      float v0 = s0[r] * 0.0625f;                    // 1/sqrt(256)
      float v1 = s1[r] * 0.0625f;
      if (key0 + m      > row) v0 = -1e30f;
      if (key0 + 16 + m > row) v1 = -1e30f;
      s_sc[(r + 8 * half) * 32 + m]      = v0;
      s_sc[(r + 8 * half) * 32 + 16 + m] = v1;
    }
    asm volatile("s_wait_dscnt 0x0" ::: "memory");

    // --- online softmax: row = m, this lane handles keys [half*16, half*16+16) ---
    {
      float vv[16], mx = -1e30f;
#pragma unroll
      for (int i = 0; i < 16; ++i) {
        vv[i] = s_sc[m * 32 + half * 16 + i];
        mx = fmaxf(mx, vv[i]);
      }
      mx = fmaxf(mx, __shfl_xor(mx, 16, 32));
      float oldm = s_m[m];
      float newm = fmaxf(oldm, mx);
      float fac  = __expf(oldm - newm);
      float sum  = 0.0f;
#pragma unroll
      for (int i = 0; i < 16; ++i) {
        float p = __expf(vv[i] - newm);
        sum += p;
        s_p[m * 32 + half * 16 + i] = f2bf(p);
      }
      sum += __shfl_xor(sum, 16, 32);
      if (half == 0) {
        s_m[m] = newm;
        s_l[m] = s_l[m] * fac + sum;
        s_f[m] = fac;
      }
    }
    asm volatile("s_wait_dscnt 0x0" ::: "memory");

    // --- rescale accumulators by per-row factor ---
    float fr[8];
#pragma unroll
    for (int r = 0; r < 8; ++r) fr[r] = s_f[r + 8 * half];
#pragma unroll
    for (int t = 0; t < 16; ++t)
#pragma unroll
      for (int r = 0; r < 8; ++r) accO[t][r] *= fr[r];

    // --- P A-fragment from LDS ---
    BF16Frag pf;
    pf.q[0] = *(const v4u*)&s_p[m * 32 + half * 16];
    pf.q[1] = *(const v4u*)&s_p[m * 32 + half * 16 + 8];

#if HAS_ASYNC_LDS
#if __has_builtin(__builtin_amdgcn_s_wait_asynccnt)
    __builtin_amdgcn_s_wait_asynccnt(0);
#else
    asm volatile("s_wait_asynccnt 0x0" ::: "memory");
#endif
#endif
    // --- O += P @ V : 16 d-tiles ---
#pragma unroll
    for (int t = 0; t < 16; ++t) {
      BF16Frag vf;
#if HAS_ASYNC_LDS
      const unsigned short* vsrc = &s_v[(t * 32 + lane) * 16];
#else
      const unsigned short* vsrc = Vb + (size_t)(t * 16 + m) * Sc + key0 + half * 16;
#endif
      vf.q[0] = *(const v4u*)(vsrc);
      vf.q[1] = *(const v4u*)(vsrc + 8);
      accO[t] = __builtin_amdgcn_wmma_f32_16x16x32_bf16(false, pf.v, false, vf.v, (short)0, accO[t], false, false);
    }
  }

  // normalize and store bf16 [row, h*D + d]
  float li[8];
#pragma unroll
  for (int r = 0; r < 8; ++r) li[r] = 1.0f / s_l[r + 8 * half];
#pragma unroll
  for (int t = 0; t < 16; ++t)
#pragma unroll
    for (int r = 0; r < 8; ++r) {
      int row = q0 + r + 8 * half;
      AO[(size_t)(b * Sc + row) * NQ + h * Dc + t * 16 + m] = f2bf(accO[t][r] * li[r]);
    }
}

// ---------- host ----------
extern "C" void kernel_launch(void* const* d_in, const int* in_sizes, int n_in,
                              void* d_out, int out_size, void* d_ws, size_t ws_size,
                              hipStream_t stream) {
  const float* hidden = (const float*)d_in[0];
  // d_in[1] = attention_mask (pure causal; applied analytically)
  const int*   pos = (const int*)d_in[2];
  const float* Wq  = (const float*)d_in[3];
  const float* Wk  = (const float*)d_in[4];
  const float* Wv  = (const float*)d_in[5];
  const float* Wo  = (const float*)d_in[6];
  const float* qnw = (const float*)d_in[7];
  const float* knw = (const float*)d_in[8];
  float* out = (float*)d_out;

  char* ws = (char*)d_ws;
  size_t off = 0;
  auto alloc = [&](size_t bytes) {
    void* p = ws + off;
    off += (bytes + 255) & ~(size_t)255;
    return p;
  };
  unsigned short* Xb  = (unsigned short*)alloc((size_t)ROWS * KIN * 2);
  unsigned short* Wqt = (unsigned short*)alloc((size_t)NQ   * KIN * 2);
  unsigned short* Wkt = (unsigned short*)alloc((size_t)NKVN * KIN * 2);
  unsigned short* Wvt = (unsigned short*)alloc((size_t)NKVN * KIN * 2);
  unsigned short* Wot = (unsigned short*)alloc((size_t)Hc   * NQ  * 2);
  float* Qf = (float*)alloc((size_t)ROWS * NQ   * 4);
  float* Kf = (float*)alloc((size_t)ROWS * NKVN * 4);
  float* Vf = (float*)alloc((size_t)ROWS * NKVN * 4);
  unsigned short* Qh = (unsigned short*)alloc((size_t)ROWS * NQ   * 2);
  unsigned short* Kh = (unsigned short*)alloc((size_t)ROWS * NKVN * 2);
  unsigned short* Vt = (unsigned short*)alloc((size_t)ROWS * NKVN * 2);
  unsigned short* AO = (unsigned short*)Qf;   // Qf is dead after rms_rope(Q)

  // 1) convert / transpose weights + activations to bf16
  {
    int n = ROWS * KIN;
    conv_bf16_kernel<<<(n + 255) / 256, 256, 0, stream>>>(hidden, Xb, n);
  }
  tconv_bf16_kernel<<<(KIN * NQ   + 255) / 256, 256, 0, stream>>>(Wq, Wqt, KIN, NQ);
  tconv_bf16_kernel<<<(KIN * NKVN + 255) / 256, 256, 0, stream>>>(Wk, Wkt, KIN, NKVN);
  tconv_bf16_kernel<<<(KIN * NKVN + 255) / 256, 256, 0, stream>>>(Wv, Wvt, KIN, NKVN);
  tconv_bf16_kernel<<<(NQ  * Hc   + 255) / 256, 256, 0, stream>>>(Wo, Wot, NQ, Hc);

  // 2) QKV projections (bf16 WMMA, f32 out)
  gemm_bf16_nt<<<dim3(ROWS / 256, NQ   / 64), 256, 0, stream>>>(Xb, Wqt, Qf, ROWS, NQ,   KIN);
  gemm_bf16_nt<<<dim3(ROWS / 256, NKVN / 64), 256, 0, stream>>>(Xb, Wkt, Kf, ROWS, NKVN, KIN);
  gemm_bf16_nt<<<dim3(ROWS / 256, NKVN / 64), 256, 0, stream>>>(Xb, Wvt, Vf, ROWS, NKVN, KIN);

  // 3) RMSNorm + RoPE (Q, K) and V transpose, all to bf16
  rms_rope_kernel<<<dim3(ROWS, NHc),  256, 0, stream>>>(Qf, qnw, pos, Qh, NHc);
  rms_rope_kernel<<<dim3(ROWS, NKVc), 256, 0, stream>>>(Kf, knw, pos, Kh, NKVc);
  vtrans_kernel<<<(ROWS * NKVN + 255) / 256, 256, 0, stream>>>(Vf, Vt);

  // 4) causal GQA flash attention
  attn_kernel<<<dim3(Sc / 16, NHc, Bc), 32, 0, stream>>>(Qh, Kh, Vt, AO);

  // 5) output projection -> f32 d_out
  gemm_bf16_nt<<<dim3(ROWS / 256, Hc / 64), 256, 0, stream>>>(AO, Wot, out, ROWS, Hc, NQ);
  (void)in_sizes; (void)n_in; (void)out_size; (void)ws_size;
}
